// SequentialLlama4TextMoe_83880711291460
// MI455X (gfx1250) — matrix operand order repaired and checked by reading, not compiled, not run on hardware
//
#include <hip/hip_runtime.h>
#include <cstdint>

// Problem constants (match reference)
#define TOKENS 8192
#define HIDDEN 2048
#define INTER  4096
#define NEXP   8

typedef unsigned short u16;
typedef __attribute__((ext_vector_type(16))) __bf16 bf16x16;
typedef __attribute__((ext_vector_type(8)))  float  v8f;

union Frag { uint4 u[2]; bf16x16 f; };

// ---- CDNA5 async global->LDS copy (ASYNCcnt path), with safe fallback ------
#if defined(__has_builtin)
#if __has_builtin(__builtin_amdgcn_global_load_async_to_lds_b128)
#define HAVE_ASYNC_COPY 1
#endif
#endif

// Exact pointee type from the builtin's signature: GCC-style int 4-vector.
typedef int v4i_g __attribute__((vector_size(16)));
typedef __attribute__((address_space(1))) v4i_g gv4i;   // global (__device__)
typedef __attribute__((address_space(3))) v4i_g lv4i;   // LDS (__shared__)

__device__ __forceinline__ void async_copy16(const void* g, void* l) {
#if defined(HAVE_ASYNC_COPY)
  __builtin_amdgcn_global_load_async_to_lds_b128(
      (gv4i*)(uintptr_t)g, (lv4i*)(uint32_t)(uintptr_t)l, 0, 0);
#else
  *(uint4*)l = *(const uint4*)g;
#endif
}

__device__ __forceinline__ void wait_async() {
#if defined(HAVE_ASYNC_COPY)
#if __has_builtin(__builtin_amdgcn_s_wait_asynccnt)
  __builtin_amdgcn_s_wait_asynccnt(0);
#else
  asm volatile("s_wait_asynccnt 0" ::: "memory");
#endif
#endif
}

// Native converts: backend emits v_cvt_pk_bf16_f32 (RNE).
__device__ __forceinline__ uint32_t pack2_bf16(float a, float b) {
  union { __bf16 h[2]; uint32_t u; } r;
  r.h[0] = (__bf16)a;
  r.h[1] = (__bf16)b;
  return r.u;
}

__device__ __forceinline__ uint2 pack4_bf16(float4 v) {
  uint2 r;
  r.x = pack2_bf16(v.x, v.y);
  r.y = pack2_bf16(v.z, v.w);
  return r;
}

__device__ __forceinline__ u16 f32_to_bf16(float x) {
  union { __bf16 h; u16 u; } r;
  r.h = (__bf16)x;
  return r.u;
}

__device__ __forceinline__ v8f wmma_bf16(const bf16x16& a, const bf16x16& b, const v8f& c) {
  // D(16x16,f32) = A(16x32,bf16) x B(32x16,bf16) + C
  return __builtin_amdgcn_wmma_f32_16x16x32_bf16(false, a, false, b, (short)0, c, false, false);
}

// ---------------------------------------------------------------- router ----
__global__ __launch_bounds__(256) void k_init(int* counts, int* cursor) {
  if (threadIdx.x < NEXP) { counts[threadIdx.x] = 0; cursor[threadIdx.x] = 0; }
}

__global__ __launch_bounds__(256) void k_router(const float* __restrict__ X,
                                                const float* __restrict__ RW,
                                                float* __restrict__ scores,
                                                int* __restrict__ eid,
                                                float* __restrict__ wgt,
                                                int* __restrict__ counts) {
  const int wid = threadIdx.x >> 5, lane = threadIdx.x & 31;
  const int t = blockIdx.x * 8 + wid;
  if (t >= TOKENS) return;
  float acc[NEXP];
#pragma unroll
  for (int e = 0; e < NEXP; ++e) acc[e] = 0.f;
  const float* xr = X + (size_t)t * HIDDEN;
  for (int k = lane; k < HIDDEN; k += 32) {
    const float xv = xr[k];
#pragma unroll
    for (int e = 0; e < NEXP; ++e) acc[e] += xv * RW[e * HIDDEN + k];
  }
#pragma unroll
  for (int e = 0; e < NEXP; ++e)
#pragma unroll
    for (int off = 16; off > 0; off >>= 1) acc[e] += __shfl_xor(acc[e], off, 32);
  if (lane == 0) {
    int best = 0; float bv = acc[0];
#pragma unroll
    for (int e = 1; e < NEXP; ++e) if (acc[e] > bv) { bv = acc[e]; best = e; }
    const float s = 1.f / (1.f + __expf(-bv));
#pragma unroll
    for (int e = 0; e < NEXP; ++e) scores[(size_t)t * NEXP + e] = (e == best) ? s : 0.f;
    eid[t] = best; wgt[t] = s;
    atomicAdd(&counts[best], 1);
  }
}

__global__ void k_prefix(const int* __restrict__ counts, int* __restrict__ offsets,
                         int* __restrict__ cursor) {
  if (threadIdx.x == 0) {
    int run = 0;
    for (int e = 0; e < NEXP; ++e) { offsets[e] = run; cursor[e] = run; run += counts[e]; }
  }
}

__global__ __launch_bounds__(256) void k_scatter(const int* __restrict__ eid,
                                                 int* __restrict__ cursor,
                                                 int* __restrict__ sorted) {
  const int t = blockIdx.x * 256 + threadIdx.x;
  if (t >= TOKENS) return;
  const int e = eid[t];
  const int pos = atomicAdd(&cursor[e], 1);
  sorted[pos] = t;
}

// ------------------------------------------------- stage A: gate/up + silu ---
// C-tile 128x64, K-step 32, double-buffered LDS (one barrier per K-step).
// 8 waves: 4 (M) x 2 (N), each wave owns a 32x32 patch for both gate and up.
__global__ __launch_bounds__(256) void k_gateup(
    const float* __restrict__ X,
    const float* __restrict__ GW, const float* __restrict__ UW,
    const float* __restrict__ SG, const float* __restrict__ SU,
    const int* __restrict__ counts, const int* __restrict__ offsets,
    const int* __restrict__ sorted, u16* __restrict__ H) {
  const int s = blockIdx.z;               // 0..7 experts, 8 = shared
  int nrows, hbase; const int* tok; const float* gw; const float* uw;
  if (s < NEXP) {
    nrows = counts[s]; tok = sorted + offsets[s];
    gw = GW + (size_t)s * INTER * HIDDEN; uw = UW + (size_t)s * INTER * HIDDEN;
    hbase = TOKENS + offsets[s];
  } else { nrows = TOKENS; tok = nullptr; gw = SG; uw = SU; hbase = 0; }

  const int m0 = blockIdx.x * 128;
  if (m0 >= nrows) return;
  const int n0 = blockIdx.y * 64;

  __shared__ u16 lA[2][128 * 32];
  __shared__ u16 lG[2][64 * 32];
  __shared__ u16 lU[2][64 * 32];

  const int tid = threadIdx.x, wid = tid >> 5, lane = tid & 31;
  const int msub = (wid & 3) * 32, nsub = (wid >> 2) * 32;
  const int l15 = lane & 15, hi = lane >> 4;

  // staging ownership: A -> 2 threads per row (64B halves); B -> 4 threads/row.
  // token gather + row guard hoisted to once per thread.
  const int arow = tid >> 1, ahalf = tid & 1;
  const int brow = tid >> 2, bq = tid & 3;
  const float* aptr = nullptr;
  if (m0 + arow < nrows) {
    const int trow = tok ? tok[m0 + arow] : (m0 + arow);
    aptr = X + (size_t)trow * HIDDEN;
  }
  const float* gptr = gw + (size_t)(n0 + brow) * HIDDEN;
  const float* uptr = uw + (size_t)(n0 + brow) * HIDDEN;

  auto stage = [&](int k0, int buf) {
#pragma unroll
    for (int i = 0; i < 4; ++i) {
      const int c4 = ahalf * 4 + i;
      uint2 pk = make_uint2(0u, 0u);
      if (aptr) pk = pack4_bf16(*(const float4*)(aptr + k0 + c4 * 4));
      *(uint2*)&lA[buf][arow * 32 + c4 * 4] = pk;
    }
#pragma unroll
    for (int i = 0; i < 2; ++i) {
      const int c4 = bq * 2 + i;
      *(uint2*)&lG[buf][brow * 32 + c4 * 4] = pack4_bf16(*(const float4*)(gptr + k0 + c4 * 4));
      *(uint2*)&lU[buf][brow * 32 + c4 * 4] = pack4_bf16(*(const float4*)(uptr + k0 + c4 * 4));
    }
    if (k0 + 32 < HIDDEN) {                  // stream-prefetch next K-slab of weights
      __builtin_prefetch(gptr + k0 + 32 + bq * 8, 0, 0);
      __builtin_prefetch(uptr + k0 + 32 + bq * 8, 0, 0);
    }
  };

  v8f accg[2][2] = {};
  v8f accu[2][2] = {};

  stage(0, 0);
  __syncthreads();

  const int NK = HIDDEN / 32;
  for (int kk = 0; kk < NK; ++kk) {
    const int cur = kk & 1;
    if (kk + 1 < NK) stage((kk + 1) * 32, cur ^ 1);   // overlap with WMMA below

    // fragments in contiguous k order (k = 16*hi + s): A 16x32 bf16, B 32x16 bf16
    Frag a[2], bg[2], bu[2];
#pragma unroll
    for (int i = 0; i < 2; ++i) {
      const int M = msub + i * 16 + l15;
      const int kb = hi * 16;                // lanes<16: K0-15; lanes>=16: K16-31
      a[i].u[0] = *(const uint4*)&lA[cur][M * 32 + kb];
      a[i].u[1] = *(const uint4*)&lA[cur][M * 32 + kb + 8];
    }
#pragma unroll
    for (int j = 0; j < 2; ++j) {
      const int N = nsub + j * 16 + l15;     // lanes<16: K0-15; lanes>=16: K16-31
      bg[j].u[0] = *(const uint4*)&lG[cur][N * 32 + hi * 16];
      bg[j].u[1] = *(const uint4*)&lG[cur][N * 32 + hi * 16 + 8];
      bu[j].u[0] = *(const uint4*)&lU[cur][N * 32 + hi * 16];
      bu[j].u[1] = *(const uint4*)&lU[cur][N * 32 + hi * 16 + 8];
    }
#pragma unroll
    for (int i = 0; i < 2; ++i)
#pragma unroll
      for (int j = 0; j < 2; ++j) {
        accg[i][j] = wmma_bf16(a[i].f, bg[j].f, accg[i][j]);
        accu[i][j] = wmma_bf16(a[i].f, bu[j].f, accu[i][j]);
      }
    __syncthreads();
  }

  // epilogue: h = silu(g) * u  -> bf16 H (compact rows)
  const int hi8 = hi * 8;
#pragma unroll
  for (int i = 0; i < 2; ++i)
#pragma unroll
    for (int v = 0; v < 8; ++v) {            // C layout: lane<16 -> M=v ; lane>=16 -> M=8+v
      const int m = m0 + msub + i * 16 + hi8 + v;
      if (m < nrows) {
        u16* hrow = H + (size_t)(hbase + m) * INTER;
#pragma unroll
        for (int j = 0; j < 2; ++j) {
          const int N = n0 + nsub + j * 16 + l15;
          const float g = accg[i][j][v], uu = accu[i][j][v];
          hrow[N] = f32_to_bf16((g / (1.f + __expf(-g))) * uu);
        }
      }
    }
}

// ------------------------------------------------- stage B: down projection --
__global__ __launch_bounds__(256) void k_down(
    const u16* __restrict__ H,
    const float* __restrict__ DW, const float* __restrict__ SD,
    const int* __restrict__ counts, const int* __restrict__ offsets,
    const int* __restrict__ sorted, const float* __restrict__ wgt,
    float* __restrict__ out, int set_base) {
  const int s = set_base + blockIdx.z;
  int nrows, hbase; const int* tok; const float* dw;
  if (s < NEXP) {
    nrows = counts[s]; tok = sorted + offsets[s];
    dw = DW + (size_t)s * HIDDEN * INTER; hbase = TOKENS + offsets[s];
  } else { nrows = TOKENS; tok = nullptr; dw = SD; hbase = 0; }

  const int m0 = blockIdx.x * 128;
  if (m0 >= nrows) return;
  const int n0 = blockIdx.y * 64;

  __shared__ u16 lA[2][128 * 32];
  __shared__ u16 lB[2][64 * 32];

  const int tid = threadIdx.x, wid = tid >> 5, lane = tid & 31;
  const int msub = (wid & 3) * 32, nsub = (wid >> 2) * 32;
  const int l15 = lane & 15, hi = lane >> 4;

  const int arow = tid >> 1, ahalf = tid & 1;
  const int brow = tid >> 2, bq = tid & 3;
  const u16* aptr = nullptr;
  if (m0 + arow < nrows) aptr = H + (size_t)(hbase + m0 + arow) * INTER;
  const float* bptr = dw + (size_t)(n0 + brow) * INTER;

  auto stage = [&](int k0, int buf) {
    // A tile is already bf16 -> async DMA straight into LDS (ASYNCcnt path).
    // Inactive tail rows keep stale LDS data; they only feed D-rows that are
    // never stored (m >= nrows), so no zero-fill is required.
    if (aptr) {
#pragma unroll
      for (int i = 0; i < 2; ++i) {
        const int c = ahalf * 2 + i;
        async_copy16(aptr + k0 + c * 8, &lA[buf][arow * 32 + c * 8]);
      }
    }
#pragma unroll
    for (int i = 0; i < 2; ++i) {
      const int c4 = bq * 2 + i;
      *(uint2*)&lB[buf][brow * 32 + c4 * 4] = pack4_bf16(*(const float4*)(bptr + k0 + c4 * 4));
    }
    if (k0 + 32 < INTER) __builtin_prefetch(bptr + k0 + 32 + bq * 8, 0, 0);
  };

  v8f acc[2][2] = {};

  stage(0, 0);
  wait_async();
  __syncthreads();

  const int NK = INTER / 32;
  for (int kk = 0; kk < NK; ++kk) {
    const int cur = kk & 1;
    if (kk + 1 < NK) stage((kk + 1) * 32, cur ^ 1);

    Frag a[2], b[2];
#pragma unroll
    for (int i = 0; i < 2; ++i) {
      const int M = msub + i * 16 + l15;
      const int kb = hi * 16;
      a[i].u[0] = *(const uint4*)&lA[cur][M * 32 + kb];
      a[i].u[1] = *(const uint4*)&lA[cur][M * 32 + kb + 8];
    }
#pragma unroll
    for (int j = 0; j < 2; ++j) {
      const int N = nsub + j * 16 + l15;
      b[j].u[0] = *(const uint4*)&lB[cur][N * 32 + hi * 16];
      b[j].u[1] = *(const uint4*)&lB[cur][N * 32 + hi * 16 + 8];
    }
#pragma unroll
    for (int i = 0; i < 2; ++i)
#pragma unroll
      for (int j = 0; j < 2; ++j)
        acc[i][j] = wmma_bf16(a[i].f, b[j].f, acc[i][j]);

    wait_async();          // own async stores to LDS done before the barrier
    __syncthreads();
  }

  // epilogue: shared pass writes, expert pass accumulates (top-1 -> no races)
  const int hi8 = hi * 8;
#pragma unroll
  for (int i = 0; i < 2; ++i)
#pragma unroll
    for (int v = 0; v < 8; ++v) {
      const int m = m0 + msub + i * 16 + hi8 + v;
      if (m < nrows) {
        const int t = tok ? tok[m] : m;
        float* prow = out + (size_t)t * HIDDEN;
        if (s < NEXP) {
          const float w = wgt[t];
#pragma unroll
          for (int j = 0; j < 2; ++j) {
            const int N = n0 + nsub + j * 16 + l15;
            prow[N] += w * acc[i][j][v];
          }
        } else {
#pragma unroll
          for (int j = 0; j < 2; ++j) {
            const int N = n0 + nsub + j * 16 + l15;
            prow[N] = acc[i][j][v];
          }
        }
      }
    }
}

// ----------------------------------------------------------------- launch ----
extern "C" void kernel_launch(void* const* d_in, const int* in_sizes, int n_in,
                              void* d_out, int out_size, void* d_ws, size_t ws_size,
                              hipStream_t stream) {
  (void)in_sizes; (void)n_in; (void)out_size; (void)ws_size;
  const float* X  = (const float*)d_in[0];
  const float* RW = (const float*)d_in[1];
  const float* GW = (const float*)d_in[2];
  const float* UW = (const float*)d_in[3];
  const float* DW = (const float*)d_in[4];
  const float* SG = (const float*)d_in[5];
  const float* SU = (const float*)d_in[6];
  const float* SD = (const float*)d_in[7];

  float* out    = (float*)d_out;
  float* scores = out + (size_t)TOKENS * HIDDEN;   // tuple: (out, router_scores)

  // workspace layout (requires ~134.4 MB):
  //   [0,32)    counts[8]      [32,64)  offsets[8]   [64,96) cursor[8]
  //   [128, +32K)  expert_id[8192]
  //   [+32K, +64K) weight[8192]
  //   [+64K, +96K) sorted[8192]
  //   [128K, 128K + 16384*4096*2)  H (bf16): rows 0..8191 shared, 8192.. experts
  char* ws = (char*)d_ws;
  int*   counts  = (int*)ws;
  int*   offsets = counts + 8;
  int*   cursor  = counts + 16;
  int*   eid     = (int*)(ws + 128);
  float* wgt     = (float*)(ws + 128 + 4 * TOKENS);
  int*   sorted  = (int*)(ws + 128 + 8 * TOKENS);
  u16*   H       = (u16*)(ws + 131072);

  k_init   <<<1, 256, 0, stream>>>(counts, cursor);
  k_router <<<TOKENS / 8, 256, 0, stream>>>(X, RW, scores, eid, wgt, counts);
  k_prefix <<<1, 1, 0, stream>>>(counts, offsets, cursor);
  k_scatter<<<TOKENS / 256, 256, 0, stream>>>(eid, cursor, sorted);

  dim3 gA(TOKENS / 128, INTER / 64, NEXP + 1);     // 8 experts + shared
  k_gateup<<<gA, 256, 0, stream>>>(X, GW, UW, SG, SU, counts, offsets, sorted, H);

  dim3 gS(TOKENS / 128, HIDDEN / 64, 1);           // shared: writes out
  k_down<<<gS, 256, 0, stream>>>(H, DW, SD, counts, offsets, sorted, wgt, out, NEXP);

  dim3 gE(TOKENS / 128, HIDDEN / 64, NEXP);        // experts: accumulate
  k_down<<<gE, 256, 0, stream>>>(H, DW, SD, counts, offsets, sorted, wgt, out, 0);
}
